// GraphormerGD_10453950399033
// MI455X (gfx1250) — matrix operand order repaired and checked by reading, not compiled, not run on hardware
//
#include <hip/hip_runtime.h>
#include <hip/hip_bf16.h>
#include <math.h>

#define DEV static __device__ __forceinline__

typedef __attribute__((ext_vector_type(16))) _Float16 v16h;
typedef __attribute__((ext_vector_type(8)))  _Float16 v8h;
typedef __attribute__((ext_vector_type(2)))  _Float16 v2h;
typedef __attribute__((ext_vector_type(8)))  float    v8f;

// ---------------------------------------------------------------------------
// WMMA helper: D = A(16x32 f16) x B(32x16 f16) + C(16x16 f32)
// ---------------------------------------------------------------------------
DEV v8f wmma16(v16h a, v16h b, v8f c) {
  return __builtin_amdgcn_wmma_f32_16x16x32_f16(false, a, false, b, (short)0, c,
                                                false, false);
}

// A-fragment (16x32, f16), per CDNA5 ISA layout:
//   lane holds row m = lane&15
//   halves 0..7  -> K = kb + koff + e      (koff = 0 lanes 0-15, 8 lanes 16-31)
//   halves 8..15 -> K = kb + 16 + koff + e
DEV v16h load_a_f32(const float* __restrict__ rowk, int lane) {
  const float* p = rowk + ((lane & 16) ? 8 : 0);
  v16h r;
#pragma unroll
  for (int e = 0; e < 8; ++e) { r[e] = (_Float16)p[e]; r[8 + e] = (_Float16)p[16 + e]; }
  return r;
}
DEV v16h load_a_h(const _Float16* __restrict__ rowk, int lane) {
  const _Float16* p = rowk + ((lane & 16) ? 16 : 0);
  v16h r;
#pragma unroll
  for (int e = 0; e < 8; ++e) { r[e] = p[e]; r[8 + e] = p[8 + e]; }
  return r;
}

// ---------------------------------------------------------------------------
// Weight pre-pack: fp32 W[Kd x Nd] -> f16 B-fragments.
// B-fragment layout (per ISA): lane holds column n = lane&15;
//   element e<8 -> K row kt*32 + koff + e, e>=8 -> kt*32 + 16 + koff + (e-8)
//   (koff = 0 lanes 0-15, 8 lanes 16-31)
// Packed index: ((kt*(Nd/16)+nt)*512 + lane*16 + e)
// ---------------------------------------------------------------------------
__global__ void pack_w_kernel(const float* __restrict__ W, int Kd, int Nd,
                              int KdPad, _Float16* __restrict__ out) {
  long idx = (long)blockIdx.x * 256 + threadIdx.x;
  long total = (long)KdPad * Nd;
  if (idx >= total) return;
  int ti     = (int)(idx >> 9);
  int within = (int)(idx & 511);
  int lane   = within >> 4;
  int e      = within & 15;
  int ntiles = Nd >> 4;
  int kt = ti / ntiles;
  int nt = ti - kt * ntiles;
  int n    = (nt << 4) + (lane & 15);
  int krow = (kt << 5) + ((lane & 16) ? 8 : 0) + ((e < 8) ? e : (16 + e - 8));
  out[idx] = (krow < Kd) ? (_Float16)W[(long)krow * Nd + n] : (_Float16)0.0f;
}

// ---------------------------------------------------------------------------
// Generic GEMM: C = act(A[M=1024,Kd] @ Wpacked[Kd,Nd] + bias [+resid])
// 256 threads = 8 waves; wave computes 16(M) x 64(N); grid (Nd/64, M/128).
// MODE 0: Q  -> f16 [B,H,N,D], *sqrt(D)=8
// MODE 1: K  -> f16 [B,H,N,D]
// MODE 2: V  -> f16 [B,H,D,N] (transposed)
// MODE 3: relu -> fp32 [M,Nd]
// MODE 4: +resid -> fp32 [M,Nd]
// ---------------------------------------------------------------------------
template <int MODE>
__global__ void __launch_bounds__(256)
gemm_kernel(const float* __restrict__ A, int Kd, int Nd,
            const _Float16* __restrict__ Wp, const float* __restrict__ bias,
            const float* __restrict__ resid, float* __restrict__ outf,
            _Float16* __restrict__ outh) {
  const int lane = threadIdx.x & 31;
  const int wave = threadIdx.x >> 5;
  const int m0 = blockIdx.y * 128 + wave * 16;
  const int n0 = blockIdx.x * 64;
  const int ntiles = Nd >> 4;

  v8f acc[4];
#pragma unroll
  for (int t = 0; t < 4; ++t) {
    float bv = bias[n0 + t * 16 + (lane & 15)];
#pragma unroll
    for (int r = 0; r < 8; ++r) acc[t][r] = bv;
  }
  const float* arow = A + (long)(m0 + (lane & 15)) * Kd;
  for (int kb = 0; kb < Kd; kb += 32) {
    __builtin_prefetch(arow + kb + 64, 0, 1);  // global_prefetch_b8
    v16h a = load_a_f32(arow + kb, lane);
#pragma unroll
    for (int t = 0; t < 4; ++t) {
      long ti = (long)(kb >> 5) * ntiles + (n0 >> 4) + t;
      v16h b = *(const v16h*)(Wp + ti * 512 + lane * 16);
      acc[t] = wmma16(a, b, acc[t]);
    }
  }
  const int hi8 = (lane & 16) ? 8 : 0;
#pragma unroll
  for (int t = 0; t < 4; ++t) {
    int n = n0 + t * 16 + (lane & 15);
#pragma unroll
    for (int r = 0; r < 8; ++r) {
      int m = m0 + hi8 + r;
      float v = acc[t][r];
      if (MODE == 0 || MODE == 1) {
        if (MODE == 0) v *= 8.0f;  // D^0.5 (source divides by d^-0.5)
        int bb = m >> 8, i = m & 255, h = n >> 6, d = n & 63;
        outh[((long)(bb * 16 + h) * 256 + i) * 64 + d] = (_Float16)v;
      } else if (MODE == 2) {
        int bb = m >> 8, i = m & 255, h = n >> 6, d = n & 63;
        outh[((long)(bb * 16 + h) * 64 + d) * 256 + i] = (_Float16)v;
      } else if (MODE == 3) {
        outf[(long)m * Nd + n] = v > 0.0f ? v : 0.0f;
      } else {
        outf[(long)m * Nd + n] = v + resid[(long)m * Nd + n];
      }
    }
  }
}

// ---------------------------------------------------------------------------
// FusedDistEncoder + bias MLP, all on WMMA.
// Block = 128 threads (4 waves), one block per (b,i); each wave does 4 j-tiles.
// Output layout [B,N,H,N]: idx = (b*256+i)*4096 + h*256 + j
// ---------------------------------------------------------------------------
__global__ void __launch_bounds__(128)
encoder_bias_kernel(const int* __restrict__ sp, const float* __restrict__ rd,
                    const float* __restrict__ table, const float* __restrict__ gmeans,
                    const float* __restrict__ gstds, const float* __restrict__ gmul,
                    const float* __restrict__ gbias, const _Float16* __restrict__ pw1,
                    const float* __restrict__ b1, const _Float16* __restrict__ pw2,
                    const float* __restrict__ b2, float* __restrict__ bias_out) {
  __shared__ float s_mean[128], s_inv[128], s_coef[128];
  __shared__ float s_h[4][16][17];
  const int tid = threadIdx.x;
  if (tid < 128) {
    float sd = fabsf(gstds[tid]) + 0.01f;
    s_mean[tid] = gmeans[tid];
    s_inv[tid]  = 1.0f / sd;
    s_coef[tid] = 0.3989423f / sd;  // 1/sqrt(2*pi)
  }
  __syncthreads();

  const int lane = tid & 31, wave = tid >> 5;
  const int bi = blockIdx.x;  // b*256 + i
  const int mloc = lane & 15;
  const int koff = (lane & 16) ? 8 : 0;
  const int hi8  = (lane & 16) ? 8 : 0;
  const float gm = gmul[0], gb = gbias[0];

  for (int qq = 0; qq < 4; ++qq) {
    const int jt = wave * 4 + qq;
    const int j = jt * 16 + mloc;            // this lane's A-matrix row
    const int spv = sp[(long)bi * 256 + j];
    const float x = gm * rd[(long)bi * 256 + j] + gb;
    const float* trow = table + (long)spv * 128;

    v8f acc;
    {
      float bv = b1[lane & 15];
#pragma unroll
      for (int r = 0; r < 8; ++r) acc[r] = bv;
    }
    for (int kb = 0; kb < 256; kb += 32) {
      v16h a;
#pragma unroll
      for (int e = 0; e < 16; ++e) {
        int c = kb + ((e < 8) ? (koff + e) : (16 + koff + (e - 8)));
        float v;
        if (c < 128) {
          v = trow[c];
        } else {
          int kk = c - 128;
          float z = (x - s_mean[kk]) * s_inv[kk];
          v = s_coef[kk] * __expf(-0.5f * z * z);
        }
        a[e] = (_Float16)v;
      }
      v16h bfrag = *(const v16h*)(pw1 + (long)(kb >> 5) * 512 + lane * 16);
      acc = wmma16(a, bfrag, acc);
    }
    // exact GeLU
#pragma unroll
    for (int r = 0; r < 8; ++r) {
      float v = acc[r];
      acc[r] = 0.5f * v * (1.0f + erff(v * 0.70710678f));
    }
    // D-layout -> plain [row][col] in LDS, then second 16x16 GEMM (K padded to 32)
#pragma unroll
    for (int r = 0; r < 8; ++r) s_h[wave][hi8 + r][lane & 15] = acc[r];
    v16h a2;
#pragma unroll
    for (int e = 0; e < 8; ++e) a2[e] = (_Float16)s_h[wave][lane & 15][koff + e];
#pragma unroll
    for (int e = 8; e < 16; ++e) a2[e] = (_Float16)0.0f;
    v16h b2f = *(const v16h*)(pw2 + lane * 16);
    v8f acc2;
    {
      float bv = b2[lane & 15];
#pragma unroll
      for (int r = 0; r < 8; ++r) acc2[r] = bv;
    }
    acc2 = wmma16(a2, b2f, acc2);
    // store: h = lane&15 (D-layout n), j = jt*16 + hi8 + r (D-layout m)
    float* dst = bias_out + (long)bi * 4096 + (lane & 15) * 256 + jt * 16 + hi8;
#pragma unroll
    for (int r = 0; r < 8; ++r) dst[r] = acc2[r];
  }
}

// ---------------------------------------------------------------------------
// Biased attention per (b,h).
//  * bias slab (16 rows x 256 cols) is DMA-staged into the LDS score buffer
//    with global_load_async_to_lds_b128 (ASYNCcnt), overlapped with Q loads
//  * scores = Q K^T accumulated on top of the staged bias (C-init from LDS)
//  * row softmax writes P as packed f16 -> PV A-fragments are 2x ds_load_b128
// 64 threads = 2 waves; each wave owns a 16-row i-tile. grid = (B*H, N/32).
// ---------------------------------------------------------------------------
__global__ void __launch_bounds__(64)
attention_kernel(const _Float16* __restrict__ q, const _Float16* __restrict__ k,
                 const _Float16* __restrict__ vt, const float* __restrict__ bias,
                 float* __restrict__ ctx) {
  // stride 260 floats: rows 16B-aligned for async b128 LDS writes, <=2-way banks
  __shared__ __attribute__((aligned(16))) float    sS[2][16][260];
  __shared__ __attribute__((aligned(16))) _Float16 sP[2][16][256];
  const int lane = threadIdx.x & 31;
  const int wave = threadIdx.x >> 5;
  const int bh = blockIdx.x;
  const int bb = bh >> 4, h = bh & 15;
  const int i0 = blockIdx.y * 32 + wave * 16;
  const int mloc = lane & 15;
  const int hi8 = (lane & 16) ? 8 : 0;
  const int ksel = (lane & 16) ? 16 : 0;

  const _Float16* qbase = q  + (long)bh * 256 * 64;
  const _Float16* kbase = k  + (long)bh * 256 * 64;
  const _Float16* vbase = vt + (long)bh * 64 * 256;
  const float* bptr = bias + (long)bb * (256 * 4096) + h * 256;

  // --- async DMA: 16 rows x 256 floats of bias -> sS (32 x b128 per wave) ---
#pragma unroll 4
  for (int it = 0; it < 32; ++it) {
    int id = it * 32 + lane;           // 1024 16-byte chunks / wave-slab
    int row = id >> 6;
    int c4 = (id & 63) << 2;
    const float* g = bptr + (long)(i0 + row) * 4096 + c4;
    unsigned l = (unsigned)(uintptr_t)&sS[wave][row][c4];
    asm volatile("global_load_async_to_lds_b128 %0, %1, off"
                 :: "v"(l), "v"(g) : "memory");
  }

  // overlap: Q fragments while the DMA is in flight
  const _Float16* qrow = qbase + (long)(i0 + mloc) * 64;
  v16h qa0 = load_a_h(qrow, lane);
  v16h qa1 = load_a_h(qrow + 32, lane);

  asm volatile("s_wait_asynccnt 0x0" ::: "memory");

  for (int jt = 0; jt < 16; ++jt) {
    const int j0 = jt * 16;
    v8f acc;
#pragma unroll
    for (int r = 0; r < 8; ++r) acc[r] = sS[wave][hi8 + r][j0 + mloc];
    v16h kf0 = *(const v16h*)(kbase + (long)(j0 + mloc) * 64 + ksel);
    v16h kf1 = *(const v16h*)(kbase + (long)(j0 + mloc) * 64 + 32 + ksel);
    acc = wmma16(qa0, kf0, acc);
    acc = wmma16(qa1, kf1, acc);
#pragma unroll
    for (int r = 0; r < 8; ++r) sS[wave][hi8 + r][j0 + mloc] = acc[r];
  }

  // Row softmax: two lanes per row (lane, lane+16), combine via shfl_xor(16).
  // Pass 2 writes P directly as packed f16 into sP (unnormalized exp).
  float invv;
  {
    const float* srow = sS[wave][mloc];
    const int c0 = (lane >> 4) * 128;
    float mx = -3.0e38f;
#pragma unroll 4
    for (int c = 0; c < 128; ++c) mx = fmaxf(mx, srow[c0 + c]);
    mx = fmaxf(mx, __shfl_xor(mx, 16, 32));
    float sum = 0.0f;
#pragma unroll 2
    for (int c = 0; c < 128; c += 2) {
      float e0 = __expf(srow[c0 + c] - mx);
      float e1 = __expf(srow[c0 + c + 1] - mx);
      sum += e0 + e1;
      v2h p2; p2[0] = (_Float16)e0; p2[1] = (_Float16)e1;
      *(v2h*)&sP[wave][mloc][c0 + c] = p2;
    }
    sum += __shfl_xor(sum, 16, 32);
    invv = 1.0f / sum;  // rows scale uniformly -> apply after P@V
  }
  float rinv[8];
#pragma unroll
  for (int r = 0; r < 8; ++r) rinv[r] = __shfl(invv, hi8 + r, 32);

  const int koff = (lane & 16) ? 8 : 0;
#pragma unroll
  for (int t = 0; t < 4; ++t) {
    const int d0 = t * 16;
    v8f acc;
#pragma unroll
    for (int r = 0; r < 8; ++r) acc[r] = 0.0f;
    for (int kb = 0; kb < 256; kb += 32) {
      v8h lo = *(const v8h*)&sP[wave][mloc][kb + ksel];
      v8h hi = *(const v8h*)&sP[wave][mloc][kb + ksel + 8];
      v16h a;
#pragma unroll
      for (int e = 0; e < 8; ++e) { a[e] = lo[e]; a[8 + e] = hi[e]; }
      v16h bf = *(const v16h*)(vbase + (long)(d0 + mloc) * 256 + kb + ksel);
      acc = wmma16(a, bf, acc);
    }
    float* cp = ctx + (long)(bb * 256 + i0 + hi8) * 1024 + h * 64 + d0 + mloc;
#pragma unroll
    for (int r = 0; r < 8; ++r) cp[(long)r * 1024] = acc[r] * rinv[r];
  }
}

// ---------------------------------------------------------------------------
// LayerNorm over F=1024; one block (256 thr) per row.
// ---------------------------------------------------------------------------
__global__ void __launch_bounds__(256)
layernorm_kernel(const float* __restrict__ x, const float* __restrict__ g,
                 const float* __restrict__ b, float* __restrict__ out) {
  __shared__ float p1[8], p2[8];
  const int row = blockIdx.x;
  const int lane = threadIdx.x & 31, wave = threadIdx.x >> 5;
  const float* xr = x + (long)row * 1024;
  float v[4], s = 0.0f;
#pragma unroll
  for (int i = 0; i < 4; ++i) { v[i] = xr[threadIdx.x + i * 256]; s += v[i]; }
#pragma unroll
  for (int off = 16; off > 0; off >>= 1) s += __shfl_xor(s, off, 32);
  if (lane == 0) p1[wave] = s;
  __syncthreads();
  float tot = 0.0f;
#pragma unroll
  for (int i = 0; i < 8; ++i) tot += p1[i];
  const float mean = tot * (1.0f / 1024.0f);
  float vs = 0.0f;
#pragma unroll
  for (int i = 0; i < 4; ++i) { float d = v[i] - mean; vs += d * d; }
#pragma unroll
  for (int off = 16; off > 0; off >>= 1) vs += __shfl_xor(vs, off, 32);
  if (lane == 0) p2[wave] = vs;
  __syncthreads();
  float vtot = 0.0f;
#pragma unroll
  for (int i = 0; i < 8; ++i) vtot += p2[i];
  const float rstd = rsqrtf(vtot * (1.0f / 1024.0f) + 1e-5f);
#pragma unroll
  for (int i = 0; i < 4; ++i) {
    int c = threadIdx.x + i * 256;
    out[(long)row * 1024 + c] = (v[i] - mean) * rstd * g[c] + b[c];
  }
}

// ---------------------------------------------------------------------------
extern "C" void kernel_launch(void* const* d_in, const int* in_sizes, int n_in,
                              void* d_out, int out_size, void* d_ws, size_t ws_size,
                              hipStream_t stream) {
  const float* nfeat  = (const float*)d_in[0];
  const int*   sp     = (const int*)  d_in[1];
  const float* rd     = (const float*)d_in[2];
  const float* table  = (const float*)d_in[3];
  const float* gmeans = (const float*)d_in[4];
  const float* gstds  = (const float*)d_in[5];
  const float* gmul   = (const float*)d_in[6];
  const float* gbias  = (const float*)d_in[7];
  const float* mlp_w1 = (const float*)d_in[8];
  const float* mlp_b1 = (const float*)d_in[9];
  const float* mlp_w2 = (const float*)d_in[10];
  const float* mlp_b2 = (const float*)d_in[11];
  const float* wq = (const float*)d_in[12]; const float* bq = (const float*)d_in[13];
  const float* wk = (const float*)d_in[14]; const float* bk = (const float*)d_in[15];
  const float* wv = (const float*)d_in[16]; const float* bv = (const float*)d_in[17];
  const float* wo = (const float*)d_in[18]; const float* bo = (const float*)d_in[19];
  const float* fw1 = (const float*)d_in[20]; const float* fb1 = (const float*)d_in[21];
  const float* fw2 = (const float*)d_in[22]; const float* fb2 = (const float*)d_in[23];
  const float* ln1g = (const float*)d_in[24]; const float* ln1b = (const float*)d_in[25];
  const float* ln2g = (const float*)d_in[26]; const float* ln2b = (const float*)d_in[27];
  float* out = (float*)d_out;

  char* wsp = (char*)d_ws;
  auto alloc = [&](size_t bytes) {
    char* p = wsp;
    wsp += (bytes + 255) & ~(size_t)255;
    return p;
  };
  const long M = 1024, F = 1024, HID = 4096;
  float*    bias_ws = (float*)   alloc(4ull * 256 * 16 * 256 * 4);  // 16 MiB [B,N,H,N]
  _Float16* q_ws    = (_Float16*)alloc(2ull * M * F);
  _Float16* k_ws    = (_Float16*)alloc(2ull * M * F);
  _Float16* v_ws    = (_Float16*)alloc(2ull * M * F);
  float*    ctx_ws  = (float*)   alloc(4ull * M * F);
  float*    tmp_ws  = (float*)   alloc(4ull * M * F);
  float*    x1_ws   = (float*)   alloc(4ull * M * F);
  float*    hid_ws  = (float*)   alloc(4ull * M * HID);
  _Float16* pk_wq   = (_Float16*)alloc(2ull * F * F);
  _Float16* pk_wk   = (_Float16*)alloc(2ull * F * F);
  _Float16* pk_wv   = (_Float16*)alloc(2ull * F * F);
  _Float16* pk_wo   = (_Float16*)alloc(2ull * F * F);
  _Float16* pk_f1   = (_Float16*)alloc(2ull * F * HID);
  _Float16* pk_f2   = (_Float16*)alloc(2ull * HID * F);
  _Float16* pk_m1   = (_Float16*)alloc(2ull * 256 * 16);
  _Float16* pk_m2   = (_Float16*)alloc(2ull * 32 * 16);

  // 1) pack all weights into WMMA B-fragment layout (f16)
  auto pack = [&](const float* W, int Kd, int Nd, int KdPad, _Float16* o) {
    long total = (long)KdPad * Nd;
    pack_w_kernel<<<dim3((unsigned)((total + 255) / 256)), dim3(256), 0, stream>>>(
        W, Kd, Nd, KdPad, o);
  };
  pack(wq, 1024, 1024, 1024, pk_wq);
  pack(wk, 1024, 1024, 1024, pk_wk);
  pack(wv, 1024, 1024, 1024, pk_wv);
  pack(wo, 1024, 1024, 1024, pk_wo);
  pack(fw1, 1024, 4096, 1024, pk_f1);
  pack(fw2, 4096, 1024, 4096, pk_f2);
  pack(mlp_w1, 256, 16, 256, pk_m1);
  pack(mlp_w2, 16, 16, 32, pk_m2);  // K padded 16->32 with zeros

  // 2) distance encoder -> attn_bias [B,N,H,N]
  encoder_bias_kernel<<<dim3(1024), dim3(128), 0, stream>>>(
      sp, rd, table, gmeans, gstds, gmul, gbias, pk_m1, mlp_b1, pk_m2, mlp_b2,
      bias_ws);

  // 3) Q/K/V projections (f16 outputs in attention-friendly layouts)
  gemm_kernel<0><<<dim3(16, 8), dim3(256), 0, stream>>>(nfeat, 1024, 1024, pk_wq, bq,
                                                        nullptr, nullptr, q_ws);
  gemm_kernel<1><<<dim3(16, 8), dim3(256), 0, stream>>>(nfeat, 1024, 1024, pk_wk, bk,
                                                        nullptr, nullptr, k_ws);
  gemm_kernel<2><<<dim3(16, 8), dim3(256), 0, stream>>>(nfeat, 1024, 1024, pk_wv, bv,
                                                        nullptr, nullptr, v_ws);

  // 4) biased attention -> ctx [B,N,F]
  attention_kernel<<<dim3(64, 8), dim3(64), 0, stream>>>(q_ws, k_ws, v_ws, bias_ws,
                                                         ctx_ws);

  // 5) output projection + residual, LN1
  gemm_kernel<4><<<dim3(16, 8), dim3(256), 0, stream>>>(ctx_ws, 1024, 1024, pk_wo, bo,
                                                        nfeat, tmp_ws, nullptr);
  layernorm_kernel<<<dim3(1024), dim3(256), 0, stream>>>(tmp_ws, ln1g, ln1b, x1_ws);

  // 6) FFN (relu), + residual, LN2 -> out
  gemm_kernel<3><<<dim3(64, 8), dim3(256), 0, stream>>>(x1_ws, 1024, 4096, pk_f1, fb1,
                                                        nullptr, hid_ws, nullptr);
  gemm_kernel<4><<<dim3(16, 8), dim3(256), 0, stream>>>(hid_ws, 4096, 1024, pk_f2, fb2,
                                                        x1_ws, tmp_ws, nullptr);
  layernorm_kernel<<<dim3(1024), dim3(256), 0, stream>>>(tmp_ws, ln2g, ln2b, out);
}
